// CLIPAttention_2594160247114
// MI455X (gfx1250) — matrix operand order repaired and checked by reading, not compiled, not run on hardware
//
#include <hip/hip_runtime.h>
#include <hip/hip_bf16.h>

// ---------------------------------------------------------------------------
// CLIP attention block (pre-LN, residual) for MI455X / gfx1250, wave32 WMMA.
// L=1024, B=8, E=1024, H=16, HD=64.  All GEMM inner math in f16 WMMA
// (16x16x32, f32 accumulate).  GEMMs stage A/W tiles in LDS via CDNA5 async
// global->LDS copies (ASYNCcnt), double-buffered against the WMMA pipeline.
// ---------------------------------------------------------------------------

#define L_TOK   1024
#define BATCH   8
#define EMB     1024
#define NHEAD   16
#define HDIM    64
#define NTOK    (L_TOK * BATCH)         // 8192
#define SCALE_Q 0.125f                  // HD^-0.5
#define LN_EPS  1e-5f

typedef _Float16 v16h __attribute__((ext_vector_type(16)));
typedef _Float16 v8h  __attribute__((ext_vector_type(8)));
typedef float    v8f  __attribute__((ext_vector_type(8)));

struct alignas(8) H4 { _Float16 x[4]; };

__device__ __forceinline__ v8f zero_v8f() {
    v8f z;
#pragma unroll
    for (int i = 0; i < 8; ++i) z[i] = 0.0f;
    return z;
}

// A-fragment (16x32 f16, M x K): lane m (m<16) holds row m, K=0..7 in halves
// 0-7 and K=16..23 in halves 8-15; lane m+16 holds K=8..15 / 24..31.
// Caller passes base = &row[k0 + (lane>=16 ? 8 : 0)].
__device__ __forceinline__ v16h load_frag_a(const _Float16* base) {
    union { v16h v; v8h h[2]; } u;
    u.h[0] = *(const v8h*)(base);
    u.h[1] = *(const v8h*)(base + 16);
    return u.v;
}

// B-fragment (32x16 f16, K x N): lane n (n<16) holds column n, K=0..7 / 16..23;
// lane n+16 holds column n, K=8..15 / 24..31.
// Caller passes base = &col_row[k0 + (lane>=16 ? 8 : 0)].
__device__ __forceinline__ v16h load_frag_b(const _Float16* base) {
    union { v16h v; v8h h[2]; } u;
    u.h[0] = *(const v8h*)(base);
    u.h[1] = *(const v8h*)(base + 16);
    return u.v;
}

// CDNA5 async global->LDS copy, 16 bytes per lane, GV addressing.
// Operands: VDST = LDS byte address VGPR, VADDR = 64-bit global address pair.
__device__ __forceinline__ void async_copy_b128(unsigned lds_addr, const void* gaddr) {
    asm volatile("global_load_async_to_lds_b128 %0, %1, off"
                 :: "v"(lds_addr), "v"(gaddr)
                 : "memory");
}

__device__ __forceinline__ void wait_async_all() {
    asm volatile("s_wait_asynccnt 0x0" ::: "memory");
}

// Generic-pointer low 32 bits of an LDS object == its LDS byte address
// (LDS aperture base has zero low word; HW truncates flat addr to LDS offset).
__device__ __forceinline__ unsigned lds_addr_of(const void* p) {
    return (unsigned)(unsigned long long)p;
}

// ---------------------------------------------------------------------------
// Kernel 1: LayerNorm over E, fp32 -> f16 output.  One block per token.
// ---------------------------------------------------------------------------
__global__ __launch_bounds__(256) void ln_kernel(const float* __restrict__ x,
                                                 const float* __restrict__ gamma,
                                                 const float* __restrict__ beta,
                                                 _Float16* __restrict__ out) {
    __shared__ float sbuf[256];
    const int t   = blockIdx.x;
    const int tid = threadIdx.x;
    const float4 v = ((const float4*)(x + (size_t)t * EMB))[tid];

    sbuf[tid] = v.x + v.y + v.z + v.w;
    __syncthreads();
    for (int o = 128; o > 0; o >>= 1) {
        if (tid < o) sbuf[tid] += sbuf[tid + o];
        __syncthreads();
    }
    const float mu = sbuf[0] * (1.0f / EMB);
    __syncthreads();

    float dx = v.x - mu, dy = v.y - mu, dz = v.z - mu, dw = v.w - mu;
    sbuf[tid] = dx * dx + dy * dy + dz * dz + dw * dw;
    __syncthreads();
    for (int o = 128; o > 0; o >>= 1) {
        if (tid < o) sbuf[tid] += sbuf[tid + o];
        __syncthreads();
    }
    const float rs = rsqrtf(sbuf[0] * (1.0f / EMB) + LN_EPS);

    const int e = tid * 4;
    H4 o4;
    o4.x[0] = (_Float16)(dx * rs * gamma[e + 0] + beta[e + 0]);
    o4.x[1] = (_Float16)(dy * rs * gamma[e + 1] + beta[e + 1]);
    o4.x[2] = (_Float16)(dz * rs * gamma[e + 2] + beta[e + 2]);
    o4.x[3] = (_Float16)(dw * rs * gamma[e + 3] + beta[e + 3]);
    *(H4*)(out + (size_t)t * EMB + e) = o4;
}

// ---------------------------------------------------------------------------
// Kernel 2: fp32 -> f16 conversion of the four weight matrices.
// ---------------------------------------------------------------------------
__global__ __launch_bounds__(256) void cvt_kernel(const float* s0, const float* s1,
                                                  const float* s2, const float* s3,
                                                  _Float16* d0, _Float16* d1,
                                                  _Float16* d2, _Float16* d3) {
    const int w = blockIdx.y;
    const float* s = (w == 0) ? s0 : (w == 1) ? s1 : (w == 2) ? s2 : s3;
    _Float16*    d = (w == 0) ? d0 : (w == 1) ? d1 : (w == 2) ? d2 : d3;
    const size_t idx = ((size_t)blockIdx.x * 256 + threadIdx.x) * 4;
    const float4 v = *(const float4*)(s + idx);
    H4 o;
    o.x[0] = (_Float16)v.x; o.x[1] = (_Float16)v.y;
    o.x[2] = (_Float16)v.z; o.x[3] = (_Float16)v.w;
    *(H4*)(d + idx) = o;
}

// ---------------------------------------------------------------------------
// Kernel 3: WMMA GEMM  C[M=8192,N=1024] = A[8192,1024] * W[1024,1024]^T
// block = 256 threads = 8 waves (4 in M, 2 in N); wave tile = 32x64;
// block tile = 128x128.  A/W k-slabs (128x32 f16) staged in LDS with
// double-buffered async global->LDS copies; fragments then ds_load'ed.
// mode 0: q -> f16 [b][h][l][hd]   (epilogue (acc+bias)*SCALE_Q)
// mode 1: k -> f16 [b][h][l][hd]
// mode 2: v -> f16 [b][h][hd][l]   (transposed for the AV B-fragment)
// mode 3: out = acc + bias + x  -> fp32 [t][E]
// ---------------------------------------------------------------------------
__global__ __launch_bounds__(256) void gemm_kernel(const _Float16* __restrict__ A,
                                                   const _Float16* __restrict__ W,
                                                   const float* __restrict__ bias,
                                                   const float* __restrict__ xres,
                                                   _Float16* __restrict__ outh,
                                                   float* __restrict__ outf,
                                                   int mode, float scale) {
    __shared__ alignas(32) _Float16 atile[2][128 * 32];   // 8 KB per buffer
    __shared__ alignas(32) _Float16 wtile[2][128 * 32];   // 8 KB per buffer

    const int tid  = threadIdx.x;
    const int lane = tid & 31;
    const int wave = tid >> 5;
    const int wm = wave & 3;            // 4 waves along M
    const int wn = wave >> 2;           // 2 waves along N
    const int m0b = blockIdx.y * 128;
    const int n0b = blockIdx.x * 128;
    const int hl = lane >> 4;           // half-wave
    const int ln = lane & 15;

    // Each thread async-copies 2x16B of the A slab and 2x16B of the W slab.
    // Chunk c (0..511): tile row = c>>2, 8-half column group = (c&3)*8.
    const int c0row = tid >> 2,        c0col = (tid & 3) * 8;
    const int c1row = (tid + 256) >> 2, c1col = ((tid + 256) & 3) * 8;

    auto stage = [&](int k0, int buf) {
        async_copy_b128(lds_addr_of(&atile[buf][c0row * 32 + c0col]),
                        A + (size_t)(m0b + c0row) * EMB + k0 + c0col);
        async_copy_b128(lds_addr_of(&atile[buf][c1row * 32 + c1col]),
                        A + (size_t)(m0b + c1row) * EMB + k0 + c1col);
        async_copy_b128(lds_addr_of(&wtile[buf][c0row * 32 + c0col]),
                        W + (size_t)(n0b + c0row) * EMB + k0 + c0col);
        async_copy_b128(lds_addr_of(&wtile[buf][c1row * 32 + c1col]),
                        W + (size_t)(n0b + c1row) * EMB + k0 + c1col);
    };

    v8f c[2][4];
#pragma unroll
    for (int mi = 0; mi < 2; ++mi)
#pragma unroll
        for (int nj = 0; nj < 4; ++nj) c[mi][nj] = zero_v8f();

    stage(0, 0);
    wait_async_all();
    __syncthreads();

    for (int k0 = 0; k0 < EMB; k0 += 32) {
        const int cur = (k0 >> 5) & 1;
        if (k0 + 32 < EMB) stage(k0 + 32, cur ^ 1);   // overlap DMA with math

        v16h a[2], b[4];
#pragma unroll
        for (int mi = 0; mi < 2; ++mi)
            a[mi] = load_frag_a(&atile[cur][(wm * 32 + mi * 16 + ln) * 32 + hl * 8]);
#pragma unroll
        for (int nj = 0; nj < 4; ++nj)
            b[nj] = load_frag_b(&wtile[cur][(wn * 64 + nj * 16 + ln) * 32 + hl * 8]);
#pragma unroll
        for (int mi = 0; mi < 2; ++mi)
#pragma unroll
            for (int nj = 0; nj < 4; ++nj)
                c[mi][nj] = __builtin_amdgcn_wmma_f32_16x16x32_f16(
                    false, a[mi], false, b[nj], (short)0, c[mi][nj], false, false);

        wait_async_all();       // this wave's copies into buf^1 landed
        __syncthreads();        // everyone done reading buf before next rewrite
    }

    // Epilogue: C layout = VGPR r -> row r (lanes 0-15) / row r+8 (lanes 16-31),
    // column = n0 + tile*16 + (lane&15).
    const int m0 = m0b + wm * 32;
    const int n0 = n0b + wn * 64;
#pragma unroll
    for (int mi = 0; mi < 2; ++mi) {
#pragma unroll
        for (int nj = 0; nj < 4; ++nj) {
            const int n = n0 + nj * 16 + ln;
#pragma unroll
            for (int r = 0; r < 8; ++r) {
                const int m = m0 + mi * 16 + hl * 8 + r;
                float val = c[mi][nj][r];
                if (mode < 3) {
                    val = (val + bias[n]) * scale;
                    const int l  = m >> 3;      // token row -> (l, b), B=8
                    const int bb = m & 7;
                    const int hh = n >> 6;      // col -> (h, d), HD=64
                    const int d  = n & 63;
                    size_t idx;
                    if (mode == 2)              // v transposed: [b][h][hd][l]
                        idx = ((size_t)(bb * NHEAD + hh) * HDIM + d) * L_TOK + l;
                    else                        // q/k: [b][h][l][hd]
                        idx = ((size_t)(bb * NHEAD + hh) * L_TOK + l) * HDIM + d;
                    outh[idx] = (_Float16)val;
                } else {
                    const size_t idx = (size_t)m * EMB + n;
                    outf[idx] = val + bias[n] + xres[idx];
                }
            }
        }
    }
}

// ---------------------------------------------------------------------------
// Kernel 4: flash attention.  One wave per 16-row q tile of one (b,h).
// block = 128 threads = 4 waves; grid = B*H*(L/16)/4 = 2048.
// Per 32-column step: 4 WMMAs for S = Q*K^T (K=HD=64), online softmax,
// P staged via LDS (C-layout -> A-layout), 4 WMMAs for O += P*V.
// K/V tiles are (b,h)-shared and L2-resident (192MB L2 holds everything).
// ---------------------------------------------------------------------------
__global__ __launch_bounds__(128) void attn_kernel(const _Float16* __restrict__ q,
                                                   const _Float16* __restrict__ k,
                                                   const _Float16* __restrict__ vT,
                                                   const float* __restrict__ mask,
                                                   _Float16* __restrict__ att) {
    __shared__ _Float16 pstage[4][16 * 32];

    const int lane = threadIdx.x & 31;
    const int wave = threadIdx.x >> 5;
    const int hl = lane >> 4;
    const int ln = lane & 15;

    const int g  = blockIdx.x * 4 + wave;       // 0 .. 8191
    const int i0 = (g & 63) << 4;               // q-tile start row
    const int hh = (g >> 6) & 15;
    const int bb = g >> 10;

    const _Float16* qb = q  + (size_t)(bb * NHEAD + hh) * L_TOK * HDIM;
    const _Float16* kb = k  + (size_t)(bb * NHEAD + hh) * L_TOK * HDIM;
    const _Float16* vb = vT + (size_t)(bb * NHEAD + hh) * HDIM * L_TOK;

    v16h qfrag[2];
#pragma unroll
    for (int ks = 0; ks < 2; ++ks)
        qfrag[ks] = load_frag_a(qb + (size_t)(i0 + ln) * HDIM + ks * 32 + hl * 8);

    float mrow[8], lrow[8];
    v8f o[4];
#pragma unroll
    for (int r = 0; r < 8; ++r) { mrow[r] = -1e30f; lrow[r] = 0.0f; }
#pragma unroll
    for (int dt = 0; dt < 4; ++dt) o[dt] = zero_v8f();

    for (int j0 = 0; j0 < L_TOK; j0 += 32) {
        // --- S = Q * K^T (two 16-col tiles, two k-steps over HD) + mask ---
        v8f s[2];
        s[0] = zero_v8f(); s[1] = zero_v8f();
#pragma unroll
        for (int jt = 0; jt < 2; ++jt)
#pragma unroll
            for (int ks = 0; ks < 2; ++ks) {
                v16h bf = load_frag_b(kb + (size_t)(j0 + jt * 16 + ln) * HDIM +
                                      ks * 32 + hl * 8);
                s[jt] = __builtin_amdgcn_wmma_f32_16x16x32_f16(
                    false, qfrag[ks], false, bf, (short)0, s[jt], false, false);
            }
#pragma unroll
        for (int jt = 0; jt < 2; ++jt)
#pragma unroll
            for (int r = 0; r < 8; ++r)
                s[jt][r] += mask[(size_t)(i0 + hl * 8 + r) * L_TOK + j0 + jt * 16 + ln];

        // --- online softmax: row max / exp / row sum (reduce across 16 lanes) ---
        float mnew[8], alpha[8], rsum[8];
#pragma unroll
        for (int r = 0; r < 8; ++r) {
            float rm = fmaxf(s[0][r], s[1][r]);
#pragma unroll
            for (int off = 1; off < 16; off <<= 1)
                rm = fmaxf(rm, __shfl_xor(rm, off, 32));
            mnew[r]  = fmaxf(mrow[r], rm);
            alpha[r] = __expf(mrow[r] - mnew[r]);
            mrow[r]  = mnew[r];
            rsum[r]  = 0.0f;
        }
#pragma unroll
        for (int jt = 0; jt < 2; ++jt)
#pragma unroll
            for (int r = 0; r < 8; ++r) {
                const float p = __expf(s[jt][r] - mnew[r]);
                rsum[r] += p;
                pstage[wave][(hl * 8 + r) * 32 + jt * 16 + ln] = (_Float16)p;
            }
#pragma unroll
        for (int r = 0; r < 8; ++r) {
            float rs = rsum[r];
#pragma unroll
            for (int off = 1; off < 16; off <<= 1)
                rs += __shfl_xor(rs, off, 32);
            lrow[r] = lrow[r] * alpha[r] + rs;
        }
#pragma unroll
        for (int dt = 0; dt < 4; ++dt)
#pragma unroll
            for (int r = 0; r < 8; ++r) o[dt][r] *= alpha[r];

        // same-wave DS ordering: wait for the P stores before re-gathering
        asm volatile("s_wait_dscnt 0x0" ::: "memory");

        // --- O += P * V ---
        v16h pa = load_frag_a(&pstage[wave][ln * 32 + hl * 8]);
#pragma unroll
        for (int dt = 0; dt < 4; ++dt) {
            v16h bv = load_frag_b(vb + (size_t)(dt * 16 + ln) * L_TOK + j0 + hl * 8);
            o[dt] = __builtin_amdgcn_wmma_f32_16x16x32_f16(
                false, pa, false, bv, (short)0, o[dt], false, false);
        }
    }

    // --- normalize and scatter to att[(l*B+b)*E + h*64 + d] as f16 ---
#pragma unroll
    for (int dt = 0; dt < 4; ++dt)
#pragma unroll
        for (int r = 0; r < 8; ++r) {
            const int i = i0 + hl * 8 + r;
            const int e = hh * HDIM + dt * 16 + ln;
            const float val = o[dt][r] / lrow[r];
            att[((size_t)i * BATCH + bb) * EMB + e] = (_Float16)val;
        }
}

// ---------------------------------------------------------------------------
// Host-side launcher.  Workspace layout (bytes):
//   ln    @ 0        (16M f16)    Wq/Wk/Wv/Wo @ 16M/18M/20M/22M (2M f16 each)
//   q     @ 24M      (16M)        k @ 40M    vT @ 56M    att @ 72M .. 88M
// ---------------------------------------------------------------------------
extern "C" void kernel_launch(void* const* d_in, const int* in_sizes, int n_in,
                              void* d_out, int out_size, void* d_ws, size_t ws_size,
                              hipStream_t stream) {
    const float* x     = (const float*)d_in[0];
    const float* amask = (const float*)d_in[1];
    const float* Wq    = (const float*)d_in[2];
    const float* bq    = (const float*)d_in[3];
    const float* Wk    = (const float*)d_in[4];
    const float* bk    = (const float*)d_in[5];
    const float* Wv    = (const float*)d_in[6];
    const float* bv    = (const float*)d_in[7];
    const float* Wo    = (const float*)d_in[8];
    const float* bo    = (const float*)d_in[9];
    const float* gamma = (const float*)d_in[10];
    const float* beta  = (const float*)d_in[11];
    float* out = (float*)d_out;

    char* w = (char*)d_ws;
    const size_t MB = 1024 * 1024;
    _Float16* ln  = (_Float16*)(w + 0);
    _Float16* wqh = (_Float16*)(w + 16 * MB);
    _Float16* wkh = (_Float16*)(w + 18 * MB);
    _Float16* wvh = (_Float16*)(w + 20 * MB);
    _Float16* woh = (_Float16*)(w + 22 * MB);
    _Float16* qh  = (_Float16*)(w + 24 * MB);
    _Float16* kh  = (_Float16*)(w + 40 * MB);
    _Float16* vth = (_Float16*)(w + 56 * MB);
    _Float16* ath = (_Float16*)(w + 72 * MB);

    // 1) LayerNorm -> f16
    ln_kernel<<<NTOK, 256, 0, stream>>>(x, gamma, beta, ln);

    // 2) weights fp32 -> f16
    cvt_kernel<<<dim3(1024, 4), 256, 0, stream>>>(Wq, Wk, Wv, Wo, wqh, wkh, wvh, woh);

    // 3) q / k / v projections (WMMA + async-LDS staging)
    dim3 ggrid(EMB / 128, NTOK / 128);
    gemm_kernel<<<ggrid, 256, 0, stream>>>(ln, wqh, bq, nullptr, qh, nullptr, 0, SCALE_Q);
    gemm_kernel<<<ggrid, 256, 0, stream>>>(ln, wkh, bk, nullptr, kh, nullptr, 1, 1.0f);
    gemm_kernel<<<ggrid, 256, 0, stream>>>(ln, wvh, bv, nullptr, vth, nullptr, 2, 1.0f);

    // 4) flash attention (WMMA)
    attn_kernel<<<(BATCH * NHEAD * (L_TOK / 16)) / 4, 128, 0, stream>>>(qh, kh, vth,
                                                                        amask, ath);

    // 5) output projection + bias + residual -> fp32 out
    gemm_kernel<<<ggrid, 256, 0, stream>>>(ath, woh, bo, x, nullptr, out, 3, 1.0f);
}
